// OptimizedAttention_46265387712890
// MI455X (gfx1250) — matrix operand order repaired and checked by reading, not compiled, not run on hardware
//
#include <hip/hip_runtime.h>
#include <hip/hip_bf16.h>

typedef _Float16 v16h __attribute__((ext_vector_type(16)));
typedef _Float16 v8h  __attribute__((ext_vector_type(8)));
typedef float    v8f  __attribute__((ext_vector_type(8)));
typedef unsigned int u32x4 __attribute__((ext_vector_type(4)));
typedef int i32x8 __attribute__((ext_vector_type(8)));
typedef int i32x4 __attribute__((ext_vector_type(4)));

// ---------------------------------------------------------------------------
// CDNA5 async global->LDS (CDNA5 ISA). LDS byte offset is
// the low 32 bits of the generic shared-memory address (flat LDS aperture).
// ---------------------------------------------------------------------------
__device__ __forceinline__ void async_load_b128(unsigned lds_off,
                                                const void* gaddr) {
  asm volatile("global_load_async_to_lds_b128 %0, %1, off"
               :: "v"(lds_off), "v"(gaddr)
               : "memory");
}
__device__ __forceinline__ void wait_async_le4() {
  asm volatile("s_wait_asynccnt 0x4" ::: "memory");
}
__device__ __forceinline__ void wait_async_0() {
  asm volatile("s_wait_asynccnt 0x0" ::: "memory");
}
__device__ __forceinline__ unsigned lds_off_of(const void* p) {
  return (unsigned)(uintptr_t)p;
}

// ---------------------------------------------------------------------------
// Tensor Data Mover: 2D f16 tile DMA global->LDS. Descriptor per
// CDNA5 ISA: group0 = {count=1, lds_addr, global_addr,
// type=2}; group1 = {data_size=2B, tensor_dim0/1, tile_dim0/1, dim0_stride};
// groups 2/3 zero (2D). Tracked by TENSORcnt. This toolchain exposes the
// 6-arg builtin (g0, g1, g2, g3, extra group, cpol).
// ---------------------------------------------------------------------------
__device__ __forceinline__ void tdm_load_2d_f16(unsigned lds_addr,
                                                const void* gptr,
                                                unsigned tile_w_elems,
                                                unsigned tile_h_rows,
                                                unsigned row_stride_elems) {
  const unsigned long long ga = (unsigned long long)(uintptr_t)gptr;
  u32x4 g0;
  g0[0] = 1u;                                   // count=1 (valid user D#)
  g0[1] = lds_addr;                             // LDS byte address
  g0[2] = (unsigned)ga;                         // global_addr[31:0]
  g0[3] = (unsigned)((ga >> 32) & 0x1FFFFFFu)   // global_addr[56:32]
          | (2u << 30);                         // type=2 ("image")
  i32x8 g1;
  g1[0] = (int)(1u << 16);                      // wg_mask=0, data_size=1 (2B)
  g1[1] = (int)(tile_w_elems << 16);            // tensor_dim0[15:0]
  g1[2] = (int)((tile_w_elems >> 16) | (tile_h_rows << 16));  // dim0 hi|dim1 lo
  g1[3] = (int)((tile_h_rows >> 16) | (tile_w_elems << 16));  // dim1 hi|tile_dim0
  g1[4] = (int)(tile_h_rows & 0xFFFFu);         // tile_dim1, tile_dim2=0
  g1[5] = (int)row_stride_elems;                // tensor_dim0_stride[31:0]
  g1[6] = 0;                                    // stride hi | dim1_stride lo
  g1[7] = 0;
  const i32x4 z4 = {};
  const i32x8 z8 = {};
  __builtin_amdgcn_tensor_load_to_lds(g0, g1, z4, z4, z8, 0);
}

// ---------------------------------------------------------------------------
// WMMA helpers (16x16x32 f16 -> f32, layouts per CDNA5 ISA)
// ---------------------------------------------------------------------------
__device__ __forceinline__ v16h frag_cat(v8h lo, v8h hi) {
  return __builtin_shufflevector(lo, hi, 0,1,2,3,4,5,6,7,8,9,10,11,12,13,14,15);
}

// A fragment (16x32, MxK): lane m = lane&15, g = lane>>4.
// e=0..7 -> k = g*8+e ; e=8..15 -> k = 16+g*8+(e-8). Two 16B LDS reads.
__device__ __forceinline__ v16h load_frag_a_lds(const _Float16* base, int ld) {
  const int lane = threadIdx.x & 31;
  const int m = lane & 15, g = lane >> 4;
  const v8h* p0 = reinterpret_cast<const v8h*>(base + m * ld + g * 8);
  const v8h* p1 = reinterpret_cast<const v8h*>(base + m * ld + 16 + g * 8);
  return frag_cat(*p0, *p1);
}

// B fragment (32x16, KxN) from N-major (Bt[n][k]) tile: n = lane&15, g = lane>>4.
// e=0..7 -> k = g*8+e ; e=8..15 -> k = 16+g*8+(e-8). Two 16B LDS reads.
__device__ __forceinline__ v16h load_frag_b_lds(const _Float16* base, int ld) {
  const int lane = threadIdx.x & 31;
  const int n = lane & 15, g = lane >> 4;
  const v8h* p0 = reinterpret_cast<const v8h*>(base + n * ld + g * 8);
  const v8h* p1 = reinterpret_cast<const v8h*>(base + n * ld + 16 + g * 8);
  return frag_cat(*p0, *p1);
}

__device__ __forceinline__ v8f wmma_f16(v16h a, v16h b, v8f c) {
  return __builtin_amdgcn_wmma_f32_16x16x32_f16(false, a, false, b, (short)0, c,
                                                false, false);
}

// ---------------------------------------------------------------------------
// f32 -> f16 conversion (8 elements/thread, n multiple of 2048)
// ---------------------------------------------------------------------------
__global__ __launch_bounds__(256) void cvt_f32_f16(const float* __restrict__ in,
                                                   _Float16* __restrict__ out,
                                                   int n) {
  const int i = (blockIdx.x * 256 + threadIdx.x) * 8;
  if (i >= n) return;
  float4 a = *reinterpret_cast<const float4*>(in + i);
  float4 b = *reinterpret_cast<const float4*>(in + i + 4);
  v8h h;
  h[0] = (_Float16)a.x; h[1] = (_Float16)a.y; h[2] = (_Float16)a.z; h[3] = (_Float16)a.w;
  h[4] = (_Float16)b.x; h[5] = (_Float16)b.y; h[6] = (_Float16)b.z; h[7] = (_Float16)b.w;
  *reinterpret_cast<v8h*>(out + i) = h;
}

// ---------------------------------------------------------------------------
// GEMM: out = X (MxK f16) @ W^T (W NxK f16) + bias(f32). 128x128 block tile,
// 8 waves (4x2), 2x4 accum frags/wave, K-step 32, double-buffered LDS filled
// by GLOBAL_LOAD_ASYNC_TO_LDS_B128 (next tile in flight behind the WMMAs).
// Store modes: 0 = row-major f32; 1 = (B,Hn,S,D) f16; 2 = (B,Hn,D,S) f16.
// ---------------------------------------------------------------------------
__global__ __launch_bounds__(256) void gemm_f16_wmma(
    const _Float16* __restrict__ X, const _Float16* __restrict__ W,
    const float* __restrict__ bias, void* __restrict__ out,
    int M, int N, int K, int mode, int Hn) {
  constexpr int TK = 32;
  __shared__ _Float16 As[2][128][TK];   // 2 x 8KB
  __shared__ _Float16 Bs[2][128][TK];   // 2 x 8KB

  const int t = threadIdx.x;
  const int wave = t >> 5, lane = t & 31;
  const int bm = blockIdx.y * 128, bn = blockIdx.x * 128;
  const int wm = (wave >> 1) * 32;
  const int wn = (wave & 1) * 64;

  // This thread's two 16B chunks of each 128x32-half tile.
  const int c0row = t >> 2, c0o = (t & 3) * 16;             // chunk t
  const int c1row = (t + 256) >> 2, c1o = c0o;              // chunk t+256
  const unsigned asB = lds_off_of(&As[0][0][0]);
  const unsigned bsB = lds_off_of(&Bs[0][0][0]);

  auto issue = [&](int buf, int kt) {
    const unsigned lb = (unsigned)buf * 8192u;
    async_load_b128(asB + lb + c0row * 64 + c0o,
                    (const char*)(X + (size_t)(bm + c0row) * K + kt) + c0o);
    async_load_b128(asB + lb + c1row * 64 + c1o,
                    (const char*)(X + (size_t)(bm + c1row) * K + kt) + c1o);
    async_load_b128(bsB + lb + c0row * 64 + c0o,
                    (const char*)(W + (size_t)(bn + c0row) * K + kt) + c0o);
    async_load_b128(bsB + lb + c1row * 64 + c1o,
                    (const char*)(W + (size_t)(bn + c1row) * K + kt) + c1o);
  };

  v8f acc[2][4] = {};
  issue(0, 0);

  for (int kt = 0; kt < K; kt += TK) {
    const int cur = (kt >> 5) & 1;
    const bool hasNext = (kt + TK) < K;
    if (hasNext) {
      issue(cur ^ 1, kt + TK);
      wait_async_le4();   // drain only current tile's 4 requests
    } else {
      wait_async_0();
    }
    __syncthreads();      // all waves' current tile visible

    v16h a[2], b[4];
#pragma unroll
    for (int mf = 0; mf < 2; ++mf)
      a[mf] = load_frag_a_lds(&As[cur][wm + mf * 16][0], TK);
#pragma unroll
    for (int nf = 0; nf < 4; ++nf)
      b[nf] = load_frag_b_lds(&Bs[cur][wn + nf * 16][0], TK);
#pragma unroll
    for (int mf = 0; mf < 2; ++mf)
#pragma unroll
      for (int nf = 0; nf < 4; ++nf)
        acc[mf][nf] = wmma_f16(a[mf], b[nf], acc[mf][nf]);

    __syncthreads();      // everyone done reading `cur` before it refills
  }

  const int g = lane >> 4, ln = lane & 15;
#pragma unroll
  for (int mf = 0; mf < 2; ++mf)
#pragma unroll
    for (int nf = 0; nf < 4; ++nf)
#pragma unroll
      for (int r = 0; r < 8; ++r) {
        const int row = bm + wm + mf * 16 + r + 8 * g;
        const int col = bn + wn + nf * 16 + ln;
        const float v = acc[mf][nf][r] + bias[col];
        if (mode == 0) {
          ((float*)out)[(size_t)row * N + col] = v;
        } else {
          const int bb = row >> 11, s = row & 2047;  // S = 2048
          const int hh = col >> 7, d = col & 127;    // D = 128
          size_t addr;
          if (mode == 1)
            addr = (((size_t)(bb * Hn + hh)) * 2048 + s) * 128 + d;
          else
            addr = (((size_t)(bb * Hn + hh)) * 128 + d) * 2048 + s;
          ((_Float16*)out)[addr] = (_Float16)v;
        }
      }
}

// ---------------------------------------------------------------------------
// RoPE (rotate_half) in-place on (B, nH, S=2048, D=128) f16 tensor.
// ---------------------------------------------------------------------------
__global__ __launch_bounds__(256) void rope_f16(_Float16* __restrict__ X,
                                                int total) {
  const int idx = blockIdx.x * 256 + threadIdx.x;
  if (idx >= total) return;
  const int i = idx & 63;
  const int s = (idx >> 6) & 2047;
  const int bh = idx >> 17;
  const float inv = __expf(-9.210340371976184f * (float)i * (1.0f / 64.0f));
  const float ang = (float)s * inv;
  float sn, cs;
  __sincosf(ang, &sn, &cs);
  _Float16* base = X + ((size_t)bh * 2048 + s) * 128;
  const float x0 = (float)base[i];
  const float x1 = (float)base[i + 64];
  base[i]      = (_Float16)(x0 * cs - x1 * sn);
  base[i + 64] = (_Float16)(x1 * cs + x0 * sn);
}

// ---------------------------------------------------------------------------
// Flash attention, GQA 32q/8kv heads. Block = (b, h, 128 q rows), 8 waves x
// 16 rows, KV streamed in 64-wide tiles (online softmax). K/V f16 tiles are
// DMA'd into LDS by the Tensor Data Mover (one wave issues 2 TDM descriptors
// per tile, s_wait_tensorcnt + barrier publishes them); P restaged through
// LDS (C->A layout).
// Qh (B,32,S,D) f16, Kh (B,8,S,D) f16, Vh (B,8,D,S) f16, Ah (B,S,32*D) f16.
// ---------------------------------------------------------------------------
__global__ __launch_bounds__(256) void flash_attn_wmma(
    const _Float16* __restrict__ Q, const _Float16* __restrict__ Kg,
    const _Float16* __restrict__ Vt, _Float16* __restrict__ O) {
  constexpr int S = 2048, D = 128, H = 32, KV = 8;
  __shared__ _Float16 Ks[64][128];    // Bt[n=s_kv][k=d]   (16KB)
  __shared__ _Float16 Vs[128][64];    // Bt[n=d][k=s_kv]   (16KB)
  __shared__ _Float16 Ps[8][16][64];  // per-wave P staging (16KB)

  const int t = threadIdx.x, wave = t >> 5, lane = t & 31;
  const int g = lane >> 4, ln = lane & 15;
  const int bid = blockIdx.x;
  const int qt = bid & 15;
  const int h = (bid >> 4) & 31;
  const int b = bid >> 9;
  const int kvh = h >> 2;
  const int q0 = qt * 128 + wave * 16;

  const float scale = 0.08838834764831845f;  // 1/sqrt(128)
  const _Float16* qbase = Q + ((size_t)(b * H + h)) * S * D;
  const _Float16* kbase = Kg + ((size_t)(b * KV + kvh)) * S * D;
  const _Float16* vbase = Vt + ((size_t)(b * KV + kvh)) * D * S;
  const unsigned ksB = lds_off_of(&Ks[0][0]);
  const unsigned vsB = lds_off_of(&Vs[0][0]);

  // Q fragments once (scaled), A-frag layout, direct from global f16.
  v16h qa[4];
#pragma unroll
  for (int kt = 0; kt < 4; ++kt) {
    v16h f;
#pragma unroll
    for (int e = 0; e < 16; ++e) {
      const int v = e >> 1;
      const int k = (v < 4 ? 0 : 16) + g * 8 + (v & 3) * 2 + (e & 1);
      f[e] = (_Float16)((float)qbase[(size_t)(q0 + ln) * D + kt * 32 + k] *
                        scale);
    }
    qa[kt] = f;
  }

  v8f of[8] = {};
  float rm[8], rl[8];
#pragma unroll
  for (int r = 0; r < 8; ++r) { rm[r] = -3.0e38f; rl[r] = 0.f; }

  for (int kv0 = 0; kv0 < S; kv0 += 64) {
    __syncthreads();  // prior tile fully consumed by all waves
    // TDM: K tile 64 rows x 128 elems (stride 128), V tile 128 rows x 64
    // elems (stride S). One wave drives the DMA engine for the block.
    if (wave == 0) {
      tdm_load_2d_f16(ksB, kbase + (size_t)kv0 * D, 128u, 64u, 128u);
      tdm_load_2d_f16(vsB, vbase + kv0, 64u, 128u, (unsigned)S);
      __builtin_amdgcn_s_wait_tensorcnt(0);
    }
    __syncthreads();

    // Scores: 16 q rows x 64 kv cols, K-dim 128 in 4 WMMA steps.
    v8f sf[4] = {};
#pragma unroll
    for (int nf = 0; nf < 4; ++nf)
#pragma unroll
      for (int kt = 0; kt < 4; ++kt) {
        v16h bk = load_frag_b_lds(&Ks[nf * 16][kt * 32], 128);
        sf[nf] = wmma_f16(qa[kt], bk, sf[nf]);
      }

    // Online softmax; row stats via shfl_xor butterflies in 16-lane halves.
    float sc[8], rs[8];
#pragma unroll
    for (int r = 0; r < 8; ++r) {
      float mx = fmaxf(fmaxf(sf[0][r], sf[1][r]), fmaxf(sf[2][r], sf[3][r]));
      mx = fmaxf(mx, __shfl_xor(mx, 1));
      mx = fmaxf(mx, __shfl_xor(mx, 2));
      mx = fmaxf(mx, __shfl_xor(mx, 4));
      mx = fmaxf(mx, __shfl_xor(mx, 8));
      const float nm = fmaxf(rm[r], mx);
      sc[r] = __expf(rm[r] - nm);
      rm[r] = nm;
      rs[r] = 0.f;
    }
#pragma unroll
    for (int nf = 0; nf < 4; ++nf)
#pragma unroll
      for (int r = 0; r < 8; ++r) {
        const float p = __expf(sf[nf][r] - rm[r]);
        rs[r] += p;
        Ps[wave][r + 8 * g][nf * 16 + ln] = (_Float16)p;
      }
#pragma unroll
    for (int r = 0; r < 8; ++r) {
      float s = rs[r];
      s += __shfl_xor(s, 1);
      s += __shfl_xor(s, 2);
      s += __shfl_xor(s, 4);
      s += __shfl_xor(s, 8);
      rl[r] = rl[r] * sc[r] + s;
    }
#pragma unroll
    for (int f = 0; f < 8; ++f)
#pragma unroll
      for (int r = 0; r < 8; ++r) of[f][r] *= sc[r];

    __syncthreads();  // order Ps stores before A-frag reloads (uniform flow)

    // O += P @ V : K-dim 64 in 2 WMMA steps, 8 d-tiles.
#pragma unroll
    for (int kt = 0; kt < 2; ++kt) {
      v16h pa = load_frag_a_lds(&Ps[wave][0][kt * 32], 64);
#pragma unroll
      for (int nf = 0; nf < 8; ++nf) {
        v16h bv = load_frag_b_lds(&Vs[nf * 16][kt * 32], 64);
        of[nf] = wmma_f16(pa, bv, of[nf]);
      }
    }
  }

  // Normalize and store f16 (B,S,H,D) = O-projection input.
#pragma unroll
  for (int nf = 0; nf < 8; ++nf)
#pragma unroll
    for (int r = 0; r < 8; ++r) {
      const int s = q0 + r + 8 * g;
      const int d = nf * 16 + ln;
      O[(((size_t)b * S + s) * H + h) * D + d] =
          (_Float16)(of[nf][r] / rl[r]);
    }
}

// ---------------------------------------------------------------------------
// Launch
// ---------------------------------------------------------------------------
extern "C" void kernel_launch(void* const* d_in, const int* in_sizes, int n_in,
                              void* d_out, int out_size, void* d_ws,
                              size_t ws_size, hipStream_t stream) {
  (void)in_sizes; (void)n_in; (void)out_size; (void)ws_size;
  const float* hs  = (const float*)d_in[0];
  const float* q_w = (const float*)d_in[1];
  const float* q_b = (const float*)d_in[2];
  const float* k_w = (const float*)d_in[3];
  const float* k_b = (const float*)d_in[4];
  const float* v_w = (const float*)d_in[5];
  const float* v_b = (const float*)d_in[6];
  const float* o_w = (const float*)d_in[7];
  const float* o_b = (const float*)d_in[8];

  constexpr int B = 2, S = 2048, HID = 4096, H = 32, KV = 8, D = 128;
  constexpr size_t NBIG = (size_t)B * S * HID;      // 16.8M (also q_w, o_w)
  constexpr size_t NSML = (size_t)KV * D * HID;     // 4.2M  (k_w, v_w)
  constexpr size_t NQ = (size_t)B * H * S * D;      // 16.8M
  constexpr size_t NK = (size_t)B * KV * S * D;     // 4.2M

  _Float16* p = (_Float16*)d_ws;
  _Float16* Xh  = p; p += NBIG;   // hidden f16 (B*S, HID)
  _Float16* qWh = p; p += NBIG;
  _Float16* kWh = p; p += NSML;
  _Float16* vWh = p; p += NSML;
  _Float16* oWh = p; p += NBIG;
  _Float16* Qh  = p; p += NQ;     // (B,H,S,D)
  _Float16* Kh  = p; p += NK;     // (B,KV,S,D)
  _Float16* Vh  = p; p += NK;     // (B,KV,D,S) transposed
  _Float16* Ah  = p; p += NQ;     // (B,S,H*D)

  const dim3 blk(256);
  // f32 -> f16 operand conversion
  cvt_f32_f16<<<NBIG / 2048, blk, 0, stream>>>(hs, Xh, (int)NBIG);
  cvt_f32_f16<<<NBIG / 2048, blk, 0, stream>>>(q_w, qWh, (int)NBIG);
  cvt_f32_f16<<<NSML / 2048, blk, 0, stream>>>(k_w, kWh, (int)NSML);
  cvt_f32_f16<<<NSML / 2048, blk, 0, stream>>>(v_w, vWh, (int)NSML);
  cvt_f32_f16<<<NBIG / 2048, blk, 0, stream>>>(o_w, oWh, (int)NBIG);
  // Projections (M = 4096, K = 4096)
  gemm_f16_wmma<<<dim3(HID / 128, (B * S) / 128), blk, 0, stream>>>(
      Xh, qWh, q_b, Qh, B * S, H * D, HID, /*mode=*/1, H);
  gemm_f16_wmma<<<dim3((KV * D) / 128, (B * S) / 128), blk, 0, stream>>>(
      Xh, kWh, k_b, Kh, B * S, KV * D, HID, /*mode=*/1, KV);
  gemm_f16_wmma<<<dim3((KV * D) / 128, (B * S) / 128), blk, 0, stream>>>(
      Xh, vWh, v_b, Vh, B * S, KV * D, HID, /*mode=*/2, KV);
  // RoPE
  rope_f16<<<(B * H * S * 64) / 256, blk, 0, stream>>>(Qh, B * H * S * 64);
  rope_f16<<<(B * KV * S * 64) / 256, blk, 0, stream>>>(Kh, B * KV * S * 64);
  // Attention
  flash_attn_wmma<<<B * H * (S / 128), blk, 0, stream>>>(Qh, Kh, Vh, Ah);
  // Output projection -> f32 d_out
  gemm_f16_wmma<<<dim3(HID / 128, (B * S) / 128), blk, 0, stream>>>(
      Ah, oWh, o_b, d_out, B * S, HID, HID, /*mode=*/0, 0);
}
